// LSTMVAE_56650618634686
// MI455X (gfx1250) — matrix operand order repaired and checked by reading, not compiled, not run on hardware
//
#include <hip/hip_runtime.h>
#include <hip/hip_bf16.h>

typedef __attribute__((ext_vector_type(16))) __bf16 v16bf;
typedef __attribute__((ext_vector_type(8)))  float  v8f;

#define HDIM  256
#define SEQ   256
#define LDIM  128
#define BTILE 16
#define LDH   264   // padded bf16 row stride for 256-wide tiles (16B-aligned, bank-spread)
#define LDZ   136   // padded bf16 row stride for 128-wide z

// d_ws bf16 element offsets
#define OFF_EWHH0 0
#define OFF_EWIH1 (1*262144)
#define OFF_EWHH1 (2*262144)
#define OFF_DWIH0 (3*262144)
#define OFF_DWHH0 (4*262144)
#define OFF_DWIH1 (5*262144)
#define OFF_DWHH1 (6*262144)
#define OFF_FCMU  (7*262144)
#define OFF_FCLV  (7*262144 + 32768)
#define OFF_DECIN (7*262144 + 2*32768)

#define OUT_MU  (1024*256)
#define OUT_LV  (1024*256 + 1024*128)

__device__ __forceinline__ float fast_sig(float x)  { return __builtin_amdgcn_rcpf(1.0f + __expf(-x)); }
__device__ __forceinline__ float fast_tanh(float x) { return 1.0f - 2.0f*__builtin_amdgcn_rcpf(1.0f + __expf(2.0f*x)); }

__device__ __forceinline__ v8f wmma_bf16(v16bf a, v16bf b, v8f c) {
  return __builtin_amdgcn_wmma_f32_16x16x32_bf16(false, a, false, b, (short)0, c, false, false);
}

// Opaque zero: a value the compiler cannot prove constant, redefined at every
// use site (volatile asm). Adding it to a weight pointer keeps the loads inside
// the time loop (no LICM hoist -> no VGPR blow-up/spills) while PRESERVING the
// pointer's global address-space provenance, so loads stay global_load_b128
// (separate LOADcnt) instead of flat_load_b128 (ties up LOADcnt+DScnt).
__device__ __forceinline__ int launder_zero() {
  int z = 0;
  asm volatile("" : "+v"(z));
  return z;
}

// A fragment (16x32 bf16) from LDS. lane<16: row=lane, K = k0+[0..7], k0+[16..23];
// lane>=16: row=lane-16, K = k0+[8..15], k0+[24..31].
__device__ __forceinline__ v16bf load_a(const __bf16* s, int ld, int k0, int lh, int hi) {
  const __bf16* p = s + lh*ld + k0 + (hi ? 8 : 0);
  union { uint4 q[2]; v16bf v; } u;
  u.q[0] = *(const uint4*)(p);
  u.q[1] = *(const uint4*)(p + 16);
  return u.v;
}

// B fragment (32x16 bf16) from global bf16 weight (row-major [N,K]).
// lane<16: col N=lane holds K=k0+[0..7], k0+[16..23]; lane>=16: N=lane-16 holds K=k0+[8..15], k0+[24..31].
__device__ __forceinline__ v16bf load_b(const __bf16* w, int ldk, int nrow, int k0, int hi) {
  const __bf16* p = w + (size_t)nrow*ldk + k0 + (hi ? 8 : 0);
  union { uint4 q[2]; v16bf v; } u;
  u.q[0] = *(const uint4*)(p);
  u.q[1] = *(const uint4*)(p + 16);
  return u.v;
}

// Accumulate the 4 gate tiles (i,f,g,o) for this wave's j-tile: acc[q] += A(sA) x W[q*256 + jtile..]^T
__device__ __forceinline__ void gemm_gates(v8f* acc, const __bf16* sA, int ldA, int K,
                                           const __bf16* W, int wave, int lh, int hi) {
  W += launder_zero();   // keep weight loads inside the time loop, stay global
  for (int k0 = 0; k0 < K; k0 += 32) {
    v16bf a = load_a(sA, ldA, k0, lh, hi);
#pragma unroll
    for (int q = 0; q < 4; ++q) {
      int nrow = q*256 + wave*16 + lh;
      v16bf b = load_b(W, K, nrow, k0, hi);
      acc[q] = wmma_bf16(a, b, acc[q]);
    }
  }
}

__device__ __forceinline__ void lstm_update(v8f* acc, float4 bias, float* cf,
                                            __bf16* sOut, int j, int mb, float* hout) {
#pragma unroll
  for (int r = 0; r < 8; ++r) {
    float iv = fast_sig (acc[0][r] + bias.x);
    float fv = fast_sig (acc[1][r] + bias.y);
    float gv = fast_tanh(acc[2][r] + bias.z);
    float ov = fast_sig (acc[3][r] + bias.w);
    float c  = fv*cf[r] + iv*gv;
    cf[r] = c;
    float h = ov*fast_tanh(c);
    hout[r] = h;
    sOut[(mb + r)*LDH + j] = (__bf16)h;
  }
}

__global__ void convert_weights_kernel(const float* a0, const float* a1, const float* a2,
                                       const float* a3, const float* a4, const float* a5,
                                       const float* a6, const float* s0, const float* s1,
                                       const float* s2, __bf16* ws) {
  const int BIG = 262144, SML = 32768;
  const int total = 7*BIG + 3*SML;
  for (int e = blockIdx.x*blockDim.x + threadIdx.x; e < total; e += gridDim.x*blockDim.x) {
    float v;
    if      (e < 1*BIG) v = a0[e];
    else if (e < 2*BIG) v = a1[e - 1*BIG];
    else if (e < 3*BIG) v = a2[e - 2*BIG];
    else if (e < 4*BIG) v = a3[e - 3*BIG];
    else if (e < 5*BIG) v = a4[e - 4*BIG];
    else if (e < 6*BIG) v = a5[e - 5*BIG];
    else if (e < 7*BIG) v = a6[e - 6*BIG];
    else if (e < 7*BIG + SML)   v = s0[e - 7*BIG];
    else if (e < 7*BIG + 2*SML) v = s1[e - 7*BIG - SML];
    else                        v = s2[e - 7*BIG - 2*SML];
    ws[e] = (__bf16)v;
  }
}

__global__ __launch_bounds__(512, 1)
void lstmvae_kernel(const float* __restrict__ x, const float* __restrict__ eps,
                    const float* __restrict__ enc_Wih0,
                    const float* __restrict__ enc_bih0, const float* __restrict__ enc_bhh0,
                    const float* __restrict__ enc_bih1, const float* __restrict__ enc_bhh1,
                    const float* __restrict__ fc_mu_b,  const float* __restrict__ fc_lv_b,
                    const float* __restrict__ dec_in_b,
                    const float* __restrict__ dec_bih0, const float* __restrict__ dec_bhh0,
                    const float* __restrict__ dec_bih1, const float* __restrict__ dec_bhh1,
                    const float* __restrict__ fc_out_w, const float* __restrict__ fc_out_b,
                    const __bf16* __restrict__ ws, float* __restrict__ out) {
  __shared__ __bf16 sHa[2][BTILE*LDH];   // layer-1 h (enc) / d1 (dec), ping-pong
  __shared__ __bf16 sHb[2][BTILE*LDH];   // layer-2 h (enc) / d2 (dec), ping-pong
  __shared__ __bf16 sZeroX[BTILE*LDH];   // zero x-input for decoder t=0
  __shared__ __bf16 sZ[BTILE*LDZ];       // latent z (bf16)
  __shared__ float  sMu[BTILE*LDIM];
  __shared__ float  sLv[BTILE*LDIM];
  __shared__ float  sXt[BTILE];          // encoder scalar input x_t
  __shared__ float  sPart[16*BTILE];     // per-wave recon partials

  const int tid  = threadIdx.x;
  const int wave = tid >> 5;
  const int lane = tid & 31;
  const int lh   = lane & 15;
  const int hi   = lane >> 4;
  const int mb   = hi * 8;               // this lane's first C/D row
  const int j    = wave*16 + lh;         // hidden column owned by this lane
  const int b0   = blockIdx.x * BTILE;

  // ---- init LDS ----
  for (int e = tid; e < BTILE*LDH; e += 512) {
    sHa[0][e] = (__bf16)0.0f;
    sHb[0][e] = (__bf16)0.0f;
    sZeroX[e] = (__bf16)0.0f;
  }
  if (tid < BTILE) sXt[tid] = x[(size_t)(b0 + tid)*SEQ + 0];

  // ---- per-lane constants ----
  const int n0 = j, n1 = 256 + j, n2 = 512 + j, n3 = 768 + j;
  float4 eb0 = make_float4(enc_bih0[n0]+enc_bhh0[n0], enc_bih0[n1]+enc_bhh0[n1],
                           enc_bih0[n2]+enc_bhh0[n2], enc_bih0[n3]+enc_bhh0[n3]);
  float4 eb1 = make_float4(enc_bih1[n0]+enc_bhh1[n0], enc_bih1[n1]+enc_bhh1[n1],
                           enc_bih1[n2]+enc_bhh1[n2], enc_bih1[n3]+enc_bhh1[n3]);
  float4 db0 = make_float4(dec_bih0[n0]+dec_bhh0[n0], dec_bih0[n1]+dec_bhh0[n1],
                           dec_bih0[n2]+dec_bhh0[n2], dec_bih0[n3]+dec_bhh0[n3]);
  float4 db1 = make_float4(dec_bih1[n0]+dec_bhh1[n0], dec_bih1[n1]+dec_bhh1[n1],
                           dec_bih1[n2]+dec_bhh1[n2], dec_bih1[n3]+dec_bhh1[n3]);
  float4 w0c = make_float4(enc_Wih0[n0], enc_Wih0[n1], enc_Wih0[n2], enc_Wih0[n3]);

  float c1f[8], c2f[8], hv[8];
#pragma unroll
  for (int r = 0; r < 8; ++r) { c1f[r] = 0.0f; c2f[r] = 0.0f; }

  __syncthreads();

  // ================= ENCODER =================
  int p = 0;
  for (int t = 0; t < SEQ; ++t) {
    // cell 0: g = x_t * Wih0[:,0] + h1 @ Whh0^T + eb0
    v8f acc[4] = {{0,0,0,0,0,0,0,0},{0,0,0,0,0,0,0,0},{0,0,0,0,0,0,0,0},{0,0,0,0,0,0,0,0}};
    gemm_gates(acc, sHa[p], LDH, HDIM, ws + OFF_EWHH0, wave, lh, hi);
#pragma unroll
    for (int r = 0; r < 8; ++r) {
      float xv = sXt[mb + r];
      acc[0][r] += xv*w0c.x; acc[1][r] += xv*w0c.y; acc[2][r] += xv*w0c.z; acc[3][r] += xv*w0c.w;
    }
    lstm_update(acc, eb0, c1f, sHa[p^1], j, mb, hv);
    __syncthreads();
    if (tid < BTILE && t + 1 < SEQ) sXt[tid] = x[(size_t)(b0 + tid)*SEQ + (t + 1)];

    // cell 1: g = h1_new @ Wih1^T + h2 @ Whh1^T + eb1
    v8f acc2[4] = {{0,0,0,0,0,0,0,0},{0,0,0,0,0,0,0,0},{0,0,0,0,0,0,0,0},{0,0,0,0,0,0,0,0}};
    gemm_gates(acc2, sHa[p^1], LDH, HDIM, ws + OFF_EWIH1, wave, lh, hi);
    gemm_gates(acc2, sHb[p],   LDH, HDIM, ws + OFF_EWHH1, wave, lh, hi);
    lstm_update(acc2, eb1, c2f, sHb[p^1], j, mb, hv);
    __syncthreads();
    p ^= 1;
  }

  // ================= mu / logvar =================
  {
    const __bf16* Wf = ws + (wave < 8 ? OFF_FCMU : OFF_FCLV) + launder_zero();
    const float*  bv = (wave < 8) ? fc_mu_b : fc_lv_b;
    float*       sML = (wave < 8) ? sMu : sLv;
    const int nt = wave & 7;
    const int n  = nt*16 + lh;
    v8f acc = {0,0,0,0,0,0,0,0};
    for (int k0 = 0; k0 < HDIM; k0 += 32) {
      v16bf a = load_a(sHb[p], LDH, k0, lh, hi);
      v16bf b = load_b(Wf, HDIM, n, k0, hi);
      acc = wmma_bf16(a, b, acc);
    }
    const float bias = bv[n];
    const size_t obase = (wave < 8) ? (size_t)OUT_MU : (size_t)OUT_LV;
#pragma unroll
    for (int r = 0; r < 8; ++r) {
      float v = acc[r] + bias;
      int m = mb + r;
      out[obase + (size_t)(b0 + m)*LDIM + n] = v;
      sML[m*LDIM + n] = v;
    }
  }
  __syncthreads();

  // z = mu + eps * exp(0.5*logvar)
  for (int e = tid; e < BTILE*LDIM; e += 512) {
    int m = e >> 7, n = e & 127;
    float zv = sMu[e] + eps[(size_t)(b0 + m)*LDIM + n] * __expf(0.5f*sLv[e]);
    sZ[m*LDZ + n] = (__bf16)zv;
  }
  __syncthreads();

  // hdec = z @ dec_in_w^T + dec_in_b ; init both decoder h states, zero c states
  {
    const __bf16* Wd = ws + OFF_DECIN + launder_zero();
    v8f acc = {0,0,0,0,0,0,0,0};
    for (int k0 = 0; k0 < LDIM; k0 += 32) {
      v16bf a = load_a(sZ, LDZ, k0, lh, hi);
      v16bf b = load_b(Wd, LDIM, j, k0, hi);
      acc = wmma_bf16(a, b, acc);
    }
    const float bias = dec_in_b[j];
#pragma unroll
    for (int r = 0; r < 8; ++r) {
      __bf16 hb = (__bf16)(acc[r] + bias);
      sHa[0][(mb + r)*LDH + j] = hb;
      sHb[0][(mb + r)*LDH + j] = hb;
      c1f[r] = 0.0f; c2f[r] = 0.0f;
    }
  }
  const float fcw = fc_out_w[j];
  const float fcb = fc_out_b[0];
  __syncthreads();

  // ================= DECODER =================
  p = 0;
  for (int t = 0; t < SEQ; ++t) {
    const __bf16* xsrc = (t == 0) ? sZeroX : sHb[p];   // x_t = d2_{t-1}
    // cell 0
    v8f acc[4] = {{0,0,0,0,0,0,0,0},{0,0,0,0,0,0,0,0},{0,0,0,0,0,0,0,0},{0,0,0,0,0,0,0,0}};
    gemm_gates(acc, xsrc,   LDH, HDIM, ws + OFF_DWIH0, wave, lh, hi);
    gemm_gates(acc, sHa[p], LDH, HDIM, ws + OFF_DWHH0, wave, lh, hi);
    lstm_update(acc, db0, c1f, sHa[p^1], j, mb, hv);
    __syncthreads();

    // cell 1
    v8f acc2[4] = {{0,0,0,0,0,0,0,0},{0,0,0,0,0,0,0,0},{0,0,0,0,0,0,0,0},{0,0,0,0,0,0,0,0}};
    gemm_gates(acc2, sHa[p^1], LDH, HDIM, ws + OFF_DWIH1, wave, lh, hi);
    gemm_gates(acc2, sHb[p],   LDH, HDIM, ws + OFF_DWHH1, wave, lh, hi);
    lstm_update(acc2, db1, c2f, sHb[p^1], j, mb, hv);

    // recon partial: d2[m][j] * fc_out_w[j], reduced over this wave's 16 columns
#pragma unroll
    for (int r = 0; r < 8; ++r) {
      float s = hv[r] * fcw;
      s += __shfl_xor(s, 1, 32);
      s += __shfl_xor(s, 2, 32);
      s += __shfl_xor(s, 4, 32);
      s += __shfl_xor(s, 8, 32);
      if (lh == 0) sPart[wave*16 + mb + r] = s;
    }
    __syncthreads();

    if (tid < BTILE) {
      float rsum = fcb;
#pragma unroll
      for (int w2 = 0; w2 < 16; ++w2) rsum += sPart[w2*16 + tid];
      out[(size_t)(b0 + tid)*SEQ + t] = rsum;
    }
    p ^= 1;
  }
}

extern "C" void kernel_launch(void* const* d_in, const int* in_sizes, int n_in,
                              void* d_out, int out_size, void* d_ws, size_t ws_size,
                              hipStream_t stream) {
  const float* x        = (const float*)d_in[0];
  const float* eps      = (const float*)d_in[1];
  const float* enc_Wih0 = (const float*)d_in[2];
  const float* enc_Whh0 = (const float*)d_in[3];
  const float* enc_bih0 = (const float*)d_in[4];
  const float* enc_bhh0 = (const float*)d_in[5];
  const float* enc_Wih1 = (const float*)d_in[6];
  const float* enc_Whh1 = (const float*)d_in[7];
  const float* enc_bih1 = (const float*)d_in[8];
  const float* enc_bhh1 = (const float*)d_in[9];
  const float* fc_mu_w  = (const float*)d_in[10];
  const float* fc_mu_b  = (const float*)d_in[11];
  const float* fc_lv_w  = (const float*)d_in[12];
  const float* fc_lv_b  = (const float*)d_in[13];
  const float* dec_in_w = (const float*)d_in[14];
  const float* dec_in_b = (const float*)d_in[15];
  const float* dec_Wih0 = (const float*)d_in[16];
  const float* dec_Whh0 = (const float*)d_in[17];
  const float* dec_bih0 = (const float*)d_in[18];
  const float* dec_bhh0 = (const float*)d_in[19];
  const float* dec_Wih1 = (const float*)d_in[20];
  const float* dec_Whh1 = (const float*)d_in[21];
  const float* dec_bih1 = (const float*)d_in[22];
  const float* dec_bhh1 = (const float*)d_in[23];
  const float* fc_out_w = (const float*)d_in[24];
  const float* fc_out_b = (const float*)d_in[25];

  __bf16* ws = (__bf16*)d_ws;

  convert_weights_kernel<<<1024, 256, 0, stream>>>(
      enc_Whh0, enc_Wih1, enc_Whh1, dec_Wih0, dec_Whh0, dec_Wih1, dec_Whh1,
      fc_mu_w, fc_lv_w, dec_in_w, ws);

  lstmvae_kernel<<<64, 512, 0, stream>>>(
      x, eps, enc_Wih0,
      enc_bih0, enc_bhh0, enc_bih1, enc_bhh1,
      fc_mu_b, fc_lv_b, dec_in_b,
      dec_bih0, dec_bhh0, dec_bih1, dec_bhh1,
      fc_out_w, fc_out_b, ws, (float*)d_out);
}
